// Attention_13159779794970
// MI455X (gfx1250) — matrix operand order repaired and checked by reading, not compiled, not run on hardware
//
#include <hip/hip_runtime.h>
#include <hip/hip_bf16.h>

// ---------------------------------------------------------------------------
// Edge-biased multi-head attention, MI455X (gfx1250, wave32, WMMA).
//
// B=2, N=512, DIM=256, H=8, DH=32, INNER=256.  edges is 536MB and dominates.
// We never materialize e = edges@We, and we read edges from HBM exactly ONCE:
// each (b,i) block streams its 512KB edge slice into a *resident* 264KB bf16
// LDS array (only possible with CDNA5's 320KB LDS), then:
//   pass1: sim  = SCALE*(q.k + edge_ij . qW[i,h,:] + q.be)   (WMMA from LDS)
//   softmax over j, entirely in LDS (sim row is 8x512 = 16KB, block-local)
//   pass2: out = attn@v + (attn@edge) @ We + be              (WMMA from LDS)
// Roofline: 536MB @ 23.3TB/s ~= 23us; ~4GFLOP of bf16 WMMA is ~10% of that.
// Streaming uses GLOBAL_LOAD_ASYNC_TO_LDS_B128 (ASYNCcnt) double-buffering.
// ---------------------------------------------------------------------------

#define B_  2
#define N_  512
#define DIM_ 256
#define H_  8
#define DH_ 32
#define INNER_ 256
#define EDGE_DIM_ 256
#define LDK 264     // bf16 row pitch for qW staging
#define LDP 520     // bf16 row pitch for 512-wide attn rows
#define LDF 260     // f32 row pitch for async staging slabs
#define LDE 258     // bf16 row pitch for resident edge array (516B = 129 dwords -> conflict-free)

// fused-kernel LDS carve (bytes)
#define SED_B  (512 * LDE * 2)        // 264192: resident edge row, bf16
#define STG_B  (2 * 16 * LDF * 4)     // 33280: async f32 staging (overlay: attn bf16 16*LDP*2)
#define SQW_B  (16 * LDK * 2)         // 8448:  qW bf16 (rows 8..15 zero)
#define SSIM_B (8 * 512 * 4)          // 16384: sim f32 (overlay: agg/fv staging 16*256*4)
#define SMEM_FUSED (SED_B + STG_B + SQW_B + SSIM_B + 32)   // 322336 < 320KB HW limit

__constant__ const float SCALE_C = 0.17677669529663688f;   // 32^-0.5

typedef __attribute__((ext_vector_type(16))) __bf16 v16bf;
typedef __attribute__((ext_vector_type(8)))  float  v8f;
typedef int ai4 __attribute__((vector_size(16)));   // 16B unit for async copies

// 16-bit A-matrix K pattern within a 32-wide K block (ISA 7.12.2, 16-bit A 16x32)
__device__ __forceinline__ int a_koff(int v, int hi) {
  return (v < 4 ? 0 : 16) + (hi ? 8 : 0) + ((v & 3) << 1);
}
// 16-bit B-matrix K pattern (same as A: lanes 0-15 K=0..7,16..23, lanes 16-31 K=8..15,24..31)
__device__ __forceinline__ int b_koff(int v, int hi) {
  return (v < 4 ? 0 : 16) + (hi ? 8 : 0) + ((v & 3) << 1);
}

// --------------------------- async copy helpers ----------------------------
#if defined(__has_builtin)
#if __has_builtin(__builtin_amdgcn_global_load_async_to_lds_b128)
#define HAVE_ASYNC_LDS 1
#endif
#endif
#ifndef HAVE_ASYNC_LDS
#define HAVE_ASYNC_LDS 0
#endif

// copy 16 bytes global -> LDS (async when available)
__device__ __forceinline__ void cp16(const float* __restrict__ g, float* l) {
#if HAVE_ASYNC_LDS
  __builtin_amdgcn_global_load_async_to_lds_b128(
      (__attribute__((address_space(1))) ai4*)const_cast<float*>(g),
      (__attribute__((address_space(3))) ai4*)l,
      0, 0);
#else
  *(float4*)l = *(const float4*)g;
#endif
}
// each wave issues exactly 4 async instructions per 16-row slab; async loads
// complete in order, so asynccnt<=4 means the previous slab has fully landed.
__device__ __forceinline__ void wait_slab_prev() {
#if HAVE_ASYNC_LDS
  asm volatile("s_wait_asynccnt 0x4" ::: "memory");
#endif
}
__device__ __forceinline__ void wait_slab_all() {
#if HAVE_ASYNC_LDS
  asm volatile("s_wait_asynccnt 0x0" ::: "memory");
#endif
}

// ---------------------------------------------------------------------------
// K1: node projections: q, k, v, qW (q folded through We per head), qbe = q.be
// ---------------------------------------------------------------------------
__global__ void k_proj(const float* __restrict__ nodes,
                       const float* __restrict__ Wq,  const float* __restrict__ bq,
                       const float* __restrict__ Wkv, const float* __restrict__ bkv,
                       const float* __restrict__ We,  const float* __restrict__ be,
                       float* __restrict__ q, float* __restrict__ kmat,
                       float* __restrict__ vmat, float* __restrict__ qW,
                       float* __restrict__ qbe) {
  const int row = blockIdx.x;                 // b*N + i
  const int t = threadIdx.x;                  // 256 threads
  __shared__ float lx[DIM_];
  __shared__ float lq[INNER_];
  lx[t] = nodes[(size_t)row * DIM_ + t];
  __syncthreads();

  float accq = bq[t];
  for (int c = 0; c < DIM_; ++c) accq += lx[c] * Wq[(size_t)c * INNER_ + t];
  lq[t] = accq;
  q[(size_t)row * INNER_ + t] = accq;

  float acck = bkv[t], accv = bkv[t + INNER_];
  for (int c = 0; c < DIM_; ++c) {
    const float xv = lx[c];
    acck += xv * Wkv[(size_t)c * 2 * INNER_ + t];
    accv += xv * Wkv[(size_t)c * 2 * INNER_ + t + INNER_];
  }
  kmat[(size_t)row * INNER_ + t] = acck;
  vmat[(size_t)row * INNER_ + t] = accv;
  __syncthreads();

  // qW[row][h][c] = sum_d q[row, h*32+d] * We[c, h*32+d]   (c = t)
  for (int h = 0; h < H_; ++h) {
    float a = 0.f;
    const float* wrow = We + (size_t)t * INNER_ + h * DH_;
    const float* qrow = lq + h * DH_;
    for (int d = 0; d < DH_; ++d) a += qrow[d] * wrow[d];
    qW[((size_t)row * H_ + h) * EDGE_DIM_ + t] = a;
  }
  if (t < H_) {
    float a = 0.f;
    for (int d = 0; d < DH_; ++d) a += lq[t * DH_ + d] * be[t * DH_ + d];
    qbe[(size_t)row * H_ + t] = a;
  }
}

// ---------------------------------------------------------------------------
// K2: sim0[b,h,i,j] = q . k  (unscaled) — one 16x16x32 bf16 WMMA per tile
// ---------------------------------------------------------------------------
__global__ void k_simqk(const float* __restrict__ q, const float* __restrict__ kmat,
                        float* __restrict__ sim0) {
  int blk = blockIdx.x;
  const int it = blk & 31; blk >>= 5;
  const int h  = blk & 7;  const int b = blk >> 3;
  const int lane = threadIdx.x & 31, wave = threadIdx.x >> 5;
  const int m = lane & 15, hi = lane >> 4;

  const float* qb = q + ((size_t)(b * N_ + it * 16)) * INNER_ + h * DH_;
  v16bf a;
#pragma unroll
  for (int v = 0; v < 8; ++v) {
    const int k = a_koff(v, hi);
    a[2 * v]     = (__bf16)qb[(size_t)m * INNER_ + k];
    a[2 * v + 1] = (__bf16)qb[(size_t)m * INNER_ + k + 1];
  }
  const float* kb = kmat + (size_t)b * N_ * INNER_ + h * DH_;
  for (int jt = wave; jt < N_ / 16; jt += 4) {
    v16bf bf;
#pragma unroll
    for (int v = 0; v < 8; ++v) {
      const int k = b_koff(v, hi);
      bf[2 * v]     = (__bf16)kb[(size_t)(jt * 16 + m) * INNER_ + k];
      bf[2 * v + 1] = (__bf16)kb[(size_t)(jt * 16 + m) * INNER_ + k + 1];
    }
    v8f c = {};
    c = __builtin_amdgcn_wmma_f32_16x16x32_bf16(false, a, false, bf, (short)0, c,
                                                false, false);
    float* sp = sim0 + (((size_t)(b * H_ + h) * N_ + it * 16)) * N_ + jt * 16;
#pragma unroll
    for (int r = 0; r < 8; ++r) sp[(size_t)(r + 8 * hi) * N_ + m] = c[r];
  }
}

// ---------------------------------------------------------------------------
// K3 (fused): one block per (b,i), 256 threads = 8 waves.
//  phase A: stream 512x256 f32 edge slice ONCE (async double-buffered 16-row
//           slabs) -> resident bf16 sed[512][LDE] in LDS
//  phase B: sim[h][j] = SCALE*(qW.edge + sim0 + qbe)  (8 WMMA chains / j-tile)
//  phase C: per-head softmax in LDS -> attn bf16 (staging overlay)
//  phase D: agg = attn@edge (LDS), fullv = attn@v (L2) via WMMA
//  phase E: ao[h*32+d] = fullv[h][h*32+d] + agg[h].We[:,h*32+d] + be
// ---------------------------------------------------------------------------
__global__ void k_edgeattn(const float* __restrict__ edges,
                           const float* __restrict__ qW,
                           const float* __restrict__ qbe,
                           const float* __restrict__ sim0,
                           const float* __restrict__ vmat,
                           const float* __restrict__ We,
                           const float* __restrict__ be,
                           float* __restrict__ ao) {
  extern __shared__ __align__(16) char smem[];
  __bf16* sed  = (__bf16*)smem;                               // [512][LDE]
  float*  stg  = (float*)(smem + SED_B);                      // 2 x [16][LDF]
  __bf16* spbf = (__bf16*)(smem + SED_B);                     // overlay: [16][LDP]
  __bf16* sqw  = (__bf16*)(smem + SED_B + STG_B);             // [16][LDK]
  float*  ssim = (float*)(smem + SED_B + STG_B + SQW_B);      // [8][512]
  float*  sstage = ssim;                                      // overlay after softmax
  float*  sqbe = (float*)(smem + SED_B + STG_B + SQW_B + SSIM_B);

  const int row = blockIdx.x;
  const int b = row / N_, i = row % N_;
  const int t = threadIdx.x;                  // 0..255
  const int lane = t & 31, wave = t >> 5;     // 8 waves
  const int m = lane & 15, hi = lane >> 4;
  const float* ep = edges + (size_t)row * N_ * EDGE_DIM_;

  // ---- phase A: stream edges once, convert to resident bf16 ----
  for (int idx = t; idx < 16 * 64; idx += 256) {       // slab 0: 16 rows x 64 x 16B
    const int r = idx >> 6, c4 = idx & 63;
    cp16(ep + (size_t)r * EDGE_DIM_ + c4 * 4, stg + r * LDF + c4 * 4);
  }
  // stage qW (8x256 -> bf16, rows 8..15 zero) + qbe while DMA is in flight
  const float* qwp = qW + (size_t)row * (H_ * EDGE_DIM_);
  for (int idx = t; idx < 16 * 256; idx += 256) {
    const int r = idx >> 8, c = idx & 255;
    sqw[r * LDK + c] = (r < H_) ? (__bf16)qwp[r * 256 + c] : (__bf16)0.f;
  }
  if (t < H_) sqbe[t] = qbe[(size_t)row * H_ + t];

  for (int s = 0; s < 32; ++s) {
    if (s + 1 < 32) {
      float* nxt = stg + ((s + 1) & 1) * 16 * LDF;
      const float* epn = ep + (size_t)(s + 1) * 16 * EDGE_DIM_;
      for (int idx = t; idx < 16 * 64; idx += 256) {
        const int r = idx >> 6, c4 = idx & 63;
        cp16(epn + (size_t)r * EDGE_DIM_ + c4 * 4, nxt + r * LDF + c4 * 4);
      }
      wait_slab_prev();
    } else {
      wait_slab_all();
    }
    __syncthreads();
    const float* src = stg + (s & 1) * 16 * LDF;       // slab s landed
    for (int idx = t; idx < 16 * 256; idx += 256) {    // f32 -> resident bf16
      const int r = idx >> 8, c = idx & 255;
      sed[(size_t)(s * 16 + r) * LDE + c] = (__bf16)src[r * LDF + c];
    }
    __syncthreads();                                   // buffer free for s+2
  }

  // ---- phase B: sim = SCALE*(qW.edge^T + qk + qbe), all from LDS ----
  for (int jt = wave; jt < 32; jt += 8) {
    v8f c = {};
#pragma unroll
    for (int kk = 0; kk < 256; kk += 32) {
      v16bf a, bfr;
#pragma unroll
      for (int v = 0; v < 8; ++v) {
        const int ka = kk + a_koff(v, hi);
        a[2 * v]     = sqw[m * LDK + ka];
        a[2 * v + 1] = sqw[m * LDK + ka + 1];
        const int kb = kk + b_koff(v, hi);              // B[k=c][n=j] = sed[j][c]
        const __bf16* p = sed + (size_t)(jt * 16 + m) * LDE + kb;
        bfr[2 * v]     = p[0];
        bfr[2 * v + 1] = p[1];
      }
      c = __builtin_amdgcn_wmma_f32_16x16x32_bf16(false, a, false, bfr, (short)0, c,
                                                  false, false);
    }
#pragma unroll
    for (int r = 0; r < 8; ++r) {
      const int hrow = r + 8 * hi;                      // rows 8..15 are zero pad
      if (hrow < H_) {
        const int j = jt * 16 + m;
        const float qk = sim0[(((size_t)(b * H_ + hrow) * N_ + i)) * N_ + j];
        ssim[hrow * 512 + j] = SCALE_C * (c[r] + qk + sqbe[hrow]);
      }
    }
  }
  __syncthreads();

  // ---- phase C: softmax per head row (wave w owns head w), attn -> bf16 ----
  for (int idx = t; idx < 8 * LDP; idx += 256)          // zero pad rows 8..15
    spbf[8 * LDP + idx] = (__bf16)0.f;
  {
    const int hrow = wave;                              // 8 waves == 8 heads
    float mx = -3.402823466e38f;
    for (int jj = lane; jj < 512; jj += 32) mx = fmaxf(mx, ssim[hrow * 512 + jj]);
    for (int o = 16; o; o >>= 1) mx = fmaxf(mx, __shfl_xor(mx, o, 32));
    float ev[16], sum = 0.f;
#pragma unroll
    for (int u = 0; u < 16; ++u) {
      ev[u] = __expf(ssim[hrow * 512 + lane + 32 * u] - mx);
      sum += ev[u];
    }
    for (int o = 16; o; o >>= 1) sum += __shfl_xor(sum, o, 32);
    const float inv = 1.f / sum;
#pragma unroll
    for (int u = 0; u < 16; ++u)
      spbf[hrow * LDP + lane + 32 * u] = (__bf16)(ev[u] * inv);
  }
  __syncthreads();

  // ---- phase D: agg = attn@edge (LDS) and fullv = attn@v (L2), WMMA ----
  const float* vp = vmat + (size_t)b * N_ * INNER_;
  v8f agg[2] = {}, fv[2] = {};
#pragma unroll 4
  for (int ks = 0; ks < 512; ks += 32) {
    v16bf a;
#pragma unroll
    for (int v = 0; v < 8; ++v) {
      const int kk = ks + a_koff(v, hi);
      a[2 * v]     = spbf[m * LDP + kk];
      a[2 * v + 1] = spbf[m * LDP + kk + 1];
    }
#pragma unroll
    for (int x = 0; x < 2; ++x) {
      const int n0 = (wave * 2 + x) * 16;
      v16bf bfr, bv;
#pragma unroll
      for (int v = 0; v < 8; ++v) {
        const int kb = ks + b_koff(v, hi);              // B[k=j][n=c]
        bfr[2 * v]     = sed[(size_t)kb * LDE + n0 + m];
        bfr[2 * v + 1] = sed[(size_t)(kb + 1) * LDE + n0 + m];
        bv[2 * v]      = (__bf16)vp[(size_t)kb * INNER_ + n0 + m];
        bv[2 * v + 1]  = (__bf16)vp[(size_t)(kb + 1) * INNER_ + n0 + m];
      }
      agg[x] = __builtin_amdgcn_wmma_f32_16x16x32_bf16(false, a, false, bfr,
                                                       (short)0, agg[x], false, false);
      fv[x]  = __builtin_amdgcn_wmma_f32_16x16x32_bf16(false, a, false, bv,
                                                       (short)0, fv[x], false, false);
    }
  }
  // stage: rows 0..7 = agg[h][c], rows 8..15 = (attn@v)[h][c]  (ssim overlay)
#pragma unroll
  for (int x = 0; x < 2; ++x) {
    const int n0 = (wave * 2 + x) * 16;
#pragma unroll
    for (int r = 0; r < 8; ++r) {
      const int hrow = r + 8 * hi;
      if (hrow < H_) {
        sstage[hrow * 256 + n0 + m]       = agg[x][r];
        sstage[(hrow + 8) * 256 + n0 + m] = fv[x][r];
      }
    }
  }
  __syncthreads();

  // ---- phase E: combine (softmax sums to 1 -> +be exactly) ----
  {
    const int h = t >> 5;                               // t = h*32 + d
    float acc = sstage[(h + 8) * 256 + t] + be[t];
    for (int c = 0; c < 256; ++c) acc += sstage[h * 256 + c] * We[(size_t)c * INNER_ + t];
    ao[(size_t)row * INNER_ + t] = acc;
  }
}

// ---------------------------------------------------------------------------
// K4: out = ao @ Wo + bo
// ---------------------------------------------------------------------------
__global__ void k_outproj(const float* __restrict__ ao, const float* __restrict__ Wo,
                          const float* __restrict__ bo, float* __restrict__ out) {
  const int row = blockIdx.x;
  const int t = threadIdx.x;
  __shared__ float lx[INNER_];
  lx[t] = ao[(size_t)row * INNER_ + t];
  __syncthreads();
  float acc = bo[t];
  for (int c = 0; c < INNER_; ++c) acc += lx[c] * Wo[(size_t)c * DIM_ + t];
  out[(size_t)row * DIM_ + t] = acc;
}

// ---------------------------------------------------------------------------
extern "C" void kernel_launch(void* const* d_in, const int* in_sizes, int n_in,
                              void* d_out, int out_size, void* d_ws, size_t ws_size,
                              hipStream_t stream) {
  (void)in_sizes; (void)n_in; (void)out_size; (void)ws_size;
  const float* nodes = (const float*)d_in[0];
  const float* edges = (const float*)d_in[1];
  const float* Wq  = (const float*)d_in[2];
  const float* bq  = (const float*)d_in[3];
  const float* Wkv = (const float*)d_in[4];
  const float* bkv = (const float*)d_in[5];
  const float* We  = (const float*)d_in[6];
  const float* be  = (const float*)d_in[7];
  const float* Wo  = (const float*)d_in[8];
  const float* bo  = (const float*)d_in[9];
  float* out = (float*)d_out;

  // workspace carve (floats): ~29.4 MB total
  float* q    = (float*)d_ws;                         // 262144
  float* kmat = q    + (size_t)B_ * N_ * INNER_;      // 262144
  float* vmat = kmat + (size_t)B_ * N_ * INNER_;      // 262144
  float* qW   = vmat + (size_t)B_ * N_ * INNER_;      // 2097152
  float* qbe  = qW   + (size_t)B_ * N_ * H_ * EDGE_DIM_;  // 8192
  float* sim0 = qbe  + (size_t)B_ * N_ * H_;          // 4194304 (qk logits)
  float* ao   = sim0 + (size_t)B_ * H_ * N_ * N_;     // 262144

  k_proj<<<B_ * N_, 256, 0, stream>>>(nodes, Wq, bq, Wkv, bkv, We, be,
                                      q, kmat, vmat, qW, qbe);
  k_simqk<<<B_ * H_ * (N_ / 16), 128, 0, stream>>>(q, kmat, sim0);
  k_edgeattn<<<B_ * N_, 256, SMEM_FUSED, stream>>>(edges, qW, qbe, sim0, vmat,
                                                   We, be, ao);
  k_outproj<<<B_ * N_, 256, 0, stream>>>(ao, Wo, bo, out);
}
